// SetAbstraction_62045097558091
// MI455X (gfx1250) — matrix operand order repaired and checked by reading, not compiled, not run on hardware
//
#include <hip/hip_runtime.h>
#include <hip/hip_bf16.h>

// ---------------------------------------------------------------------------
// Types for CDNA5 WMMA (wave32, 16x16x32 f16 -> f32)
// ---------------------------------------------------------------------------
typedef _Float16 v16h __attribute__((ext_vector_type(16)));
typedef _Float16 v8h  __attribute__((ext_vector_type(8)));
typedef float    v8f  __attribute__((ext_vector_type(8)));

#define N_PTS   50000
#define N_BALLS 2048
#define KNN     64
#define RADIUS2 0.04f
#define NBINS   4096

__device__ __forceinline__ v8f wmma_f16(v16h a, v16h b, v8f c) {
    // (neg_a, A, neg_b, B, c_mod, C, reuse_a, reuse_b)
    return __builtin_amdgcn_wmma_f32_16x16x32_f16(false, a, false, b,
                                                  (short)0, c, false, false);
}

// A fragment: 16x32 (MxK) f16 tile, row-major, leading dim `ld` (halves).
// lane L: M = L%16; halves 0..7 = K koff..koff+7, halves 8..15 = K koff+16..+23,
// koff = 0 (lanes 0-15) / 8 (lanes 16-31).   [ISA 7.12.2]
__device__ __forceinline__ v16h load_fragA(const _Float16* base, int ld, int lane) {
    int r    = lane & 15;
    int koff = (lane < 16) ? 0 : 8;
    const _Float16* p = base + r * ld + koff;
    v8h lo = *(const v8h*)(p);
    v8h hi = *(const v8h*)(p + 16);
    v16h out;
#pragma unroll
    for (int i = 0; i < 8; ++i) { out[i] = lo[i]; out[i + 8] = hi[i]; }
    return out;
}

// B fragment: 32x16 (KxN) tile, stored as [N][K] row-major (ld halves).
// lane L: N = L%16; halves 0..7 = K koff..koff+7, halves 8..15 = K koff+16..+23, koff = 0 / 8.
__device__ __forceinline__ v16h load_fragB(const _Float16* base, int ld, int lane) {
    int r    = lane & 15;
    int koff = (lane < 16) ? 0 : 8;
    const _Float16* p = base + r * ld + koff;
    v8h lo = *(const v8h*)(p);
    v8h hi = *(const v8h*)(p + 16);
    v16h out;
#pragma unroll
    for (int i = 0; i < 8; ++i) { out[i] = lo[i]; out[i + 8] = hi[i]; }
    return out;
}

// Epilogue for the A=weights / B=activations mapping:
// D element (M = n_out = r + 8*(lane>=16), N = m = lane%16).
// Each lane stores 8 *consecutive* output features for one point -> 1x b128.
// `base` points at tile origin [m0][n0] of a [.. ][ld] f16 activation array,
// `bias` points at bias + n0.
__device__ __forceinline__ void store_tile_f16(_Float16* base, int ld, int lane,
                                               v8f acc, const float* bias,
                                               bool relu) {
    int m  = lane & 15;
    int nb = (lane < 16) ? 0 : 8;
    v8h out;
#pragma unroll
    for (int r = 0; r < 8; ++r) {
        float v = acc[r] + bias[nb + r];
        if (relu) v = v > 0.0f ? v : 0.0f;
        out[r] = (_Float16)v;
    }
    *(v8h*)(base + m * ld + nb) = out;
}

// ---------------------------------------------------------------------------
// Kernel 0: convert/transpose/pad weights to f16 [N][K] layouts (done once).
// ---------------------------------------------------------------------------
__global__ void prep_weights_kernel(const float* __restrict__ lw1,
                                    const float* __restrict__ lw2,
                                    const float* __restrict__ lw3,
                                    const float* __restrict__ gw1,
                                    const float* __restrict__ gw2,
                                    _Float16* __restrict__ w1t,   // [64][32]
                                    _Float16* __restrict__ w2t,   // [64][64]
                                    _Float16* __restrict__ w3t,   // [128][64]
                                    _Float16* __restrict__ g1t,   // [256][128]
                                    _Float16* __restrict__ g2t) { // [512][256]
    int gid    = blockIdx.x * blockDim.x + threadIdx.x;
    int stride = gridDim.x * blockDim.x;
    for (int t = gid; t < 64 * 32; t += stride) {   // lw1: (6,64) -> [o][i] pad 6->32
        int o = t >> 5, i = t & 31;
        w1t[t] = (_Float16)((i < 6) ? lw1[i * 64 + o] : 0.0f);
    }
    for (int t = gid; t < 64 * 64; t += stride) {   // lw2: (64,64)
        int o = t >> 6, i = t & 63;
        w2t[t] = (_Float16)lw2[i * 64 + o];
    }
    for (int t = gid; t < 128 * 64; t += stride) {  // lw3: (64,128)
        int o = t >> 6, i = t & 63;
        w3t[t] = (_Float16)lw3[i * 128 + o];
    }
    for (int t = gid; t < 256 * 128; t += stride) { // gw1: (128,256)
        int o = t >> 7, i = t & 127;
        g1t[t] = (_Float16)gw1[i * 256 + o];
    }
    for (int t = gid; t < 512 * 256; t += stride) { // gw2: (256,512)
        int o = t >> 8, i = t & 255;
        g2t[t] = (_Float16)gw2[i * 512 + o];
    }
}

// ---------------------------------------------------------------------------
// Kernel 1: farthest point sampling (sequential; 1 block of 1024 = 32 waves)
// ---------------------------------------------------------------------------
__global__ void __launch_bounds__(1024)
fps_kernel(const float* __restrict__ verts,
           float* __restrict__ mind2,       // [N_PTS] workspace
           int* __restrict__ idxs,          // [N_BALLS]
           float* __restrict__ cent) {      // [N_BALLS][3]
    const int tid = threadIdx.x;
    const float3* __restrict__ v3p = (const float3*)verts;
    __shared__ float swd[32];
    __shared__ int   swi[32];
    __shared__ int   s_next;

    for (int i = tid; i < N_PTS; i += 1024) mind2[i] = 3.0e38f;
    __syncthreads();

    int last = 0;
    for (int step = 0; step < N_BALLS; ++step) {
        if (tid == 0) idxs[step] = last;
        if (tid < 3)  cent[step * 3 + tid] = verts[last * 3 + tid];

        float3 lp = v3p[last];

        float bestd = -1.0f;
        int   besti = 0x7fffffff;
        for (int i = tid; i < N_PTS; i += 1024) {
            float3 p = v3p[i];
            float dx = p.x - lp.x, dy = p.y - lp.y, dz = p.z - lp.z;
            float d2 = dx * dx + dy * dy + dz * dz;
            float m  = mind2[i];
            m = fminf(m, d2);
            mind2[i] = m;
            if (m > bestd || (m == bestd && i < besti)) { bestd = m; besti = i; }
        }
        // wave32 argmax reduction (tie -> lowest index, matches jnp.argmax)
#pragma unroll
        for (int off = 16; off > 0; off >>= 1) {
            float od = __shfl_xor(bestd, off, 32);
            int   oi = __shfl_xor(besti, off, 32);
            if (od > bestd || (od == bestd && oi < besti)) { bestd = od; besti = oi; }
        }
        if ((tid & 31) == 0) { swd[tid >> 5] = bestd; swi[tid >> 5] = besti; }
        __syncthreads();
        if (tid < 32) {
            float bd = swd[tid];
            int   bi = swi[tid];
#pragma unroll
            for (int off = 16; off > 0; off >>= 1) {
                float od = __shfl_xor(bd, off, 32);
                int   oi = __shfl_xor(bi, off, 32);
                if (od > bd || (od == bd && oi < bi)) { bd = od; bi = oi; }
            }
            if (tid == 0) s_next = bi;
        }
        __syncthreads();
        last = s_next;
    }
}

// ---------------------------------------------------------------------------
// Kernel 2: ball query — min(64, #within-radius) nearest via LDS d2 histogram
// one block (256 thr) per centroid
// ---------------------------------------------------------------------------
__global__ void __launch_bounds__(256)
ballq_kernel(const float* __restrict__ verts,
             const float* __restrict__ cent,
             int* __restrict__ nbr,          // [N_BALLS][KNN]
             int* __restrict__ cnt) {        // [N_BALLS]
    const int b   = blockIdx.x;
    const int tid = threadIdx.x;
    const float3* __restrict__ v3p = (const float3*)verts;
    __shared__ int   hist[NBINS];
    __shared__ float sc[3];
    __shared__ int   sBstar, sTarget, sCursor;

    for (int i = tid; i < NBINS; i += 256) hist[i] = 0;
    if (tid < 3) sc[tid] = cent[b * 3 + tid];
    __syncthreads();
    const float cx = sc[0], cy = sc[1], cz = sc[2];
    const float scale = (float)NBINS / RADIUS2;

    for (int i = tid; i < N_PTS; i += 256) {
        float3 p = v3p[i];
        float dx = p.x - cx, dy = p.y - cy, dz = p.z - cz;
        float d2 = dx * dx + dy * dy + dz * dz;
        if (d2 < RADIUS2) {
            int bin = (int)(d2 * scale); bin = bin < NBINS - 1 ? bin : NBINS - 1;
            atomicAdd(&hist[bin], 1);
        }
    }
    __syncthreads();
    if (tid == 0) {
        int total = 0;
        for (int i = 0; i < NBINS; ++i) total += hist[i];
        int target = total < KNN ? total : KNN;
        int cum = 0, bstar = NBINS - 1;
        for (int i = 0; i < NBINS; ++i) { cum += hist[i]; if (cum >= target) { bstar = i; break; } }
        sBstar = bstar; sTarget = target; sCursor = 0;
    }
    __syncthreads();
    const int bstar = sBstar, target = sTarget;
    int* out = nbr + (size_t)b * KNN;

    for (int i = tid; i < N_PTS; i += 256) {   // strictly inside boundary bin
        float3 p = v3p[i];
        float dx = p.x - cx, dy = p.y - cy, dz = p.z - cz;
        float d2 = dx * dx + dy * dy + dz * dz;
        if (d2 < RADIUS2) {
            int bin = (int)(d2 * scale); bin = bin < NBINS - 1 ? bin : NBINS - 1;
            if (bin < bstar) { int p2 = atomicAdd(&sCursor, 1); out[p2] = i; }
        }
    }
    __syncthreads();
    for (int i = tid; i < N_PTS; i += 256) {   // fill from boundary bin
        float3 p = v3p[i];
        float dx = p.x - cx, dy = p.y - cy, dz = p.z - cz;
        float d2 = dx * dx + dy * dy + dz * dz;
        if (d2 < RADIUS2) {
            int bin = (int)(d2 * scale); bin = bin < NBINS - 1 ? bin : NBINS - 1;
            if (bin == bstar) { int p2 = atomicAdd(&sCursor, 1); if (p2 < target) out[p2] = i; }
        }
    }
    if (tid == 0) cnt[b] = target;
}

// ---------------------------------------------------------------------------
// Kernel 3: per-centroid local MLP (6->64->64->128) via WMMA + masked max-agg
// Mapping: D[n][m] = W[n][k] X[k][m]; A = weights [n][k], B = activations [m][k].
// one block (256 thr = 8 waves) per centroid
// ---------------------------------------------------------------------------
__global__ void __launch_bounds__(256)
local_mlp_kernel(const float* __restrict__ verts,
                 const float* __restrict__ cent,
                 const int* __restrict__ nbr,
                 const int* __restrict__ cnt,
                 const _Float16* __restrict__ w1t, const float* __restrict__ lb1,
                 const _Float16* __restrict__ w2t, const float* __restrict__ lb2,
                 const _Float16* __restrict__ w3t, const float* __restrict__ lb3,
                 _Float16* __restrict__ m16) {    // [N_BALLS][128] f16
    const int b    = blockIdx.x;
    const int tid  = threadIdx.x;
    const int lane = tid & 31;
    const int w    = tid >> 5;

    __shared__ _Float16 feat[64 * 32];   // [point][K pad 6->32]
    __shared__ _Float16 h1[64 * 64];
    __shared__ _Float16 h2[64 * 64];
    __shared__ float    scv[3];
    __shared__ int      scount;

    if (tid == 0) scount = cnt[b];
    if (tid < 3)  scv[tid] = cent[b * 3 + tid];
    __syncthreads();
    const int count = scount;

    if (tid < 64) {
        _Float16* row = feat + tid * 32;
        v8h zv = {};
        if (tid < count) {
            int i = nbr[(size_t)b * KNN + tid];
            float3 p = ((const float3*)verts)[i];
            v8h r0 = zv;
            r0[0] = (_Float16)p.x;            r0[1] = (_Float16)p.y;            r0[2] = (_Float16)p.z;
            r0[3] = (_Float16)(p.x - scv[0]); r0[4] = (_Float16)(p.y - scv[1]); r0[5] = (_Float16)(p.z - scv[2]);
            *(v8h*)(row + 0) = r0;
        } else {
            *(v8h*)(row + 0) = zv;
        }
        *(v8h*)(row + 8)  = zv;
        *(v8h*)(row + 16) = zv;
        *(v8h*)(row + 24) = zv;
    }
    __syncthreads();

    // ---- layer 1: 4 n-tiles x 4 m-tiles; wave w: nt = w>>1, mt = (w&1)*2+j ----
    {
        int nt = w >> 1;
        v16h aw = load_fragA(w1t + nt * 16 * 32, 32, lane);
#pragma unroll
        for (int j = 0; j < 2; ++j) {
            int mt = (w & 1) * 2 + j;
            v8f acc = {};
            acc = wmma_f16(aw, load_fragB(feat + mt * 16 * 32, 32, lane), acc);
            store_tile_f16(h1 + mt * 16 * 64 + nt * 16, 64, lane, acc, lb1 + nt * 16, true);
        }
    }
    __syncthreads();

    // ---- layer 2: 4x4 tiles, K=64 (2 k-steps); weight frags hoisted ----
    {
        int nt = w >> 1;
        v16h aw0 = load_fragA(w2t + nt * 16 * 64 + 0,  64, lane);
        v16h aw1 = load_fragA(w2t + nt * 16 * 64 + 32, 64, lane);
#pragma unroll
        for (int j = 0; j < 2; ++j) {
            int mt = (w & 1) * 2 + j;
            v8f acc = {};
            acc = wmma_f16(aw0, load_fragB(h1 + mt * 16 * 64 + 0,  64, lane), acc);
            acc = wmma_f16(aw1, load_fragB(h1 + mt * 16 * 64 + 32, 64, lane), acc);
            store_tile_f16(h2 + mt * 16 * 64 + nt * 16, 64, lane, acc, lb2 + nt * 16, true);
        }
    }
    __syncthreads();

    // ---- layer 3: wave w owns n-tile w (16 of 128 feats), loops 4 m-tiles,
    //      fused masked max over points (cross-lane: m = lane%16) ----
    {
        int nt = w;
        v16h aw0 = load_fragA(w3t + nt * 16 * 64 + 0,  64, lane);
        v16h aw1 = load_fragA(w3t + nt * 16 * 64 + 32, 64, lane);
        int nb = (lane < 16) ? 0 : 8;
        float bias[8];
#pragma unroll
        for (int r = 0; r < 8; ++r) bias[r] = lb3[nt * 16 + nb + r];
        float mx[8];
#pragma unroll
        for (int r = 0; r < 8; ++r) mx[r] = -1e30f;

        for (int mt = 0; mt < 4; ++mt) {
            v8f acc = {};
            acc = wmma_f16(aw0, load_fragB(h2 + mt * 16 * 64 + 0,  64, lane), acc);
            acc = wmma_f16(aw1, load_fragB(h2 + mt * 16 * 64 + 32, 64, lane), acc);
            int  m     = mt * 16 + (lane & 15);
            bool valid = m < count;
#pragma unroll
            for (int r = 0; r < 8; ++r) {
                float v = acc[r] + bias[r];
                if (valid) mx[r] = fmaxf(mx[r], v);
            }
        }
        // max over the 16 points per half-wave (offsets < 16 stay in each half)
#pragma unroll
        for (int off = 1; off < 16; off <<= 1) {
#pragma unroll
            for (int r = 0; r < 8; ++r)
                mx[r] = fmaxf(mx[r], __shfl_xor(mx[r], off, 32));
        }
        if ((lane & 15) == 0) {
            v8h outv;
#pragma unroll
            for (int r = 0; r < 8; ++r) outv[r] = (_Float16)mx[r];
            *(v8h*)(m16 + (size_t)b * 128 + nt * 16 + nb) = outv;
        }
    }
}

// ---------------------------------------------------------------------------
// Kernel 4: global MLP (128->256 relu ->512) as WMMA GEMM; 64 rows per block
// ---------------------------------------------------------------------------
__global__ void __launch_bounds__(256)
global_mlp_kernel(const _Float16* __restrict__ m16,
                  const _Float16* __restrict__ g1t, const float* __restrict__ gb1,
                  const _Float16* __restrict__ g2t, const float* __restrict__ gb2,
                  float* __restrict__ out) {       // [N_BALLS][512]
    const int row0 = blockIdx.x * 64;
    const int tid  = threadIdx.x;
    const int lane = tid & 31;
    const int w    = tid >> 5;

    __shared__ _Float16 h[64 * 256];   // 32 KB
    const _Float16* A = m16 + (size_t)row0 * 128;

    // ---- g1: 16 n-tiles x 4 m-tiles, K=128 (4 k-steps); 2 n-tiles per wave ----
#pragma unroll
    for (int nn = 0; nn < 2; ++nn) {
        int nt = w * 2 + nn;
        v16h aw[4];
#pragma unroll
        for (int ks = 0; ks < 4; ++ks)
            aw[ks] = load_fragA(g1t + nt * 16 * 128 + ks * 32, 128, lane);
#pragma unroll
        for (int mt = 0; mt < 4; ++mt) {
            v8f acc = {};
#pragma unroll
            for (int ks = 0; ks < 4; ++ks)
                acc = wmma_f16(aw[ks], load_fragB(A + mt * 16 * 128 + ks * 32, 128, lane), acc);
            store_tile_f16(h + mt * 16 * 256 + nt * 16, 256, lane, acc, gb1 + nt * 16, true);
        }
    }
    __syncthreads();

    // ---- g2: 32 n-tiles x 4 m-tiles, K=256 (8 k-steps); 4 n-tiles per wave ----
    for (int nn = 0; nn < 4; ++nn) {
        int nt = w * 4 + nn;
        v16h aw[8];
#pragma unroll
        for (int ks = 0; ks < 8; ++ks)
            aw[ks] = load_fragA(g2t + nt * 16 * 256 + ks * 32, 256, lane);
        int nb = (lane < 16) ? 0 : 8;
        float bias[8];
#pragma unroll
        for (int r = 0; r < 8; ++r) bias[r] = gb2[nt * 16 + nb + r];
#pragma unroll
        for (int mt = 0; mt < 4; ++mt) {
            v8f acc = {};
#pragma unroll
            for (int ks = 0; ks < 8; ++ks)
                acc = wmma_f16(aw[ks], load_fragB(h + mt * 16 * 256 + ks * 32, 256, lane), acc);
            int m = mt * 16 + (lane & 15);
            float* op = out + (size_t)(row0 + m) * 512 + nt * 16 + nb;
#pragma unroll
            for (int r = 0; r < 8; ++r) op[r] = acc[r] + bias[r];
        }
    }
}

// ---------------------------------------------------------------------------
// Host-side launcher
// ---------------------------------------------------------------------------
extern "C" void kernel_launch(void* const* d_in, const int* in_sizes, int n_in,
                              void* d_out, int out_size, void* d_ws, size_t ws_size,
                              hipStream_t stream) {
    (void)in_sizes; (void)n_in; (void)out_size; (void)ws_size;

    const float* verts = (const float*)d_in[0];
    const float* lw1   = (const float*)d_in[1];
    const float* lb1   = (const float*)d_in[2];
    const float* lw2   = (const float*)d_in[3];
    const float* lb2   = (const float*)d_in[4];
    const float* lw3   = (const float*)d_in[5];
    const float* lb3   = (const float*)d_in[6];
    const float* gw1   = (const float*)d_in[7];
    const float* gb1   = (const float*)d_in[8];
    const float* gw2   = (const float*)d_in[9];
    const float* gb2   = (const float*)d_in[10];

    uint8_t* base = (uint8_t*)d_ws;
    size_t off = 0;
    auto alloc = [&](size_t bytes) -> void* {
        void* r = base + off;
        off += (bytes + 255) & ~(size_t)255;
        return r;
    };
    float*     mind2 = (float*)    alloc((size_t)N_PTS * 4);
    int*       idxs  = (int*)      alloc((size_t)N_BALLS * 4);
    float*     cent  = (float*)    alloc((size_t)N_BALLS * 3 * 4);
    int*       nbr   = (int*)      alloc((size_t)N_BALLS * KNN * 4);
    int*       cnt   = (int*)      alloc((size_t)N_BALLS * 4);
    _Float16*  m16   = (_Float16*) alloc((size_t)N_BALLS * 128 * 2);
    _Float16*  w1t   = (_Float16*) alloc((size_t)64 * 32 * 2);
    _Float16*  w2t   = (_Float16*) alloc((size_t)64 * 64 * 2);
    _Float16*  w3t   = (_Float16*) alloc((size_t)128 * 64 * 2);
    _Float16*  g1t   = (_Float16*) alloc((size_t)256 * 128 * 2);
    _Float16*  g2t   = (_Float16*) alloc((size_t)512 * 256 * 2);

    prep_weights_kernel<<<256, 256, 0, stream>>>(lw1, lw2, lw3, gw1, gw2,
                                                 w1t, w2t, w3t, g1t, g2t);
    fps_kernel<<<1, 1024, 0, stream>>>(verts, mind2, idxs, cent);
    ballq_kernel<<<N_BALLS, 256, 0, stream>>>(verts, cent, nbr, cnt);
    local_mlp_kernel<<<N_BALLS, 256, 0, stream>>>(verts, cent, nbr, cnt,
                                                  w1t, lb1, w2t, lb2, w3t, lb3, m16);
    global_mlp_kernel<<<N_BALLS / 64, 256, 0, stream>>>(m16, g1t, gb1, g2t, gb2,
                                                        (float*)d_out);
}
